// SeedBBoxHead_7155415515498
// MI455X (gfx1250) — matrix operand order repaired and checked by reading, not compiled, not run on hardware
//
#include <hip/hip_runtime.h>
#include <hip/hip_bf16.h>

typedef __attribute__((ext_vector_type(16))) _Float16 v16h;
typedef __attribute__((ext_vector_type(8)))  _Float16 v8h;
typedef __attribute__((ext_vector_type(8)))  float    v8f;

#define NPTS        1000000
#define NTILES      62500        // NPTS / 16
#define BSEG        512
#define DH          128

// workspace byte offsets (all 256-aligned)
#define OFF_SUMS    0            // 512*128 f32 = 262144
#define OFF_SUMS2   262144       // 262144
#define OFF_MAXU    524288       // 262144 (ordered-uint encoded float max)
#define OFF_CNT     786432       // 512 f32 = 2048
#define OFF_W1P     788480       // 96*128  f16 = 24576
#define OFF_W2P     813056       // 128*128 f16 = 32768
#define OFF_W3P     845824       // 384*256 f16 = 196608
#define OFF_W4P     1042432      // 256*128 f16 = 65536
#define OFF_W5P     1107968      // 128*16  f16 = 4096

__device__ __forceinline__ float silu_f(float x) {
    return x / (1.0f + __expf(-x));
}

// ordered-uint encoding so unsigned atomicMax == float max
__device__ __forceinline__ unsigned fkey(float v) {
    unsigned u = __float_as_uint(v);
    return (u & 0x80000000u) ? ~u : (u | 0x80000000u);
}
__device__ __forceinline__ float fkey_dec(unsigned k) {
    if (k == 0u) return 0.0f;
    return (k & 0x80000000u) ? __uint_as_float(k & 0x7fffffffu)
                             : __uint_as_float(~k);
}

// ---------------- prep: zero accumulators ----------------
__global__ void zero_ws_kernel(float* sums, float* sums2, unsigned* maxu, float* counts) {
    int i = blockIdx.x * blockDim.x + threadIdx.x;
    if (i < BSEG * DH) { sums[i] = 0.0f; sums2[i] = 0.0f; maxu[i] = 0u; }
    if (i < BSEG) counts[i] = 0.0f;
}

// ---------------- prep: pack weight (K x Ncols f32, row-major) into WMMA B layout f16 ----------------
// flat index: (((nt*KT + kt)*32 + lane)*16 + e) ; element e <-> k = kt*32 + (lane/16)*8 + (e%8) + (e/8)*16 ; col = nt*16 + lane%16
__global__ void pack_w_kernel(const float* __restrict__ W, _Float16* __restrict__ Wp,
                              int K, int Ncols, int KT, int NT) {
    int i = blockIdx.x * blockDim.x + threadIdx.x;
    int total = KT * NT * 512;
    if (i >= total) return;
    int e    = i & 15;
    int rest = i >> 4;
    int lane = rest & 31;
    int blk  = rest >> 5;          // nt*KT + kt
    int kt   = blk % KT;
    int nt   = blk / KT;
    int k    = kt * 32 + (lane >> 4) * 8 + (e & 7) + (e >> 3) * 16;
    int c    = nt * 16 + (lane & 15);
    float v  = (k < K && c < Ncols) ? W[(size_t)k * Ncols + c] : 0.0f;
    Wp[i] = (_Float16)v;
}

// ---------------- main: per-point embed MLP (WMMA) + segment pooling ----------------
__global__ __launch_bounds__(256) void embed_pool_kernel(
    const float* __restrict__ feats, const float* __restrict__ centers,
    const int* __restrict__ bidx,
    const _Float16* __restrict__ W1p, const float* __restrict__ b1,
    const _Float16* __restrict__ W2p, const float* __restrict__ b2,
    float* __restrict__ sums, float* __restrict__ sums2,
    unsigned* __restrict__ maxu, float* __restrict__ counts)
{
    __shared__ alignas(16) _Float16 sW1[3 * 8 * 512];   // 24KB
    __shared__ alignas(16) _Float16 sW2[4 * 8 * 512];   // 32KB
    __shared__ alignas(16) _Float16 sH[8][16 * 128];    // 32KB, per-wave h1 tile

    // cooperative weight load into LDS (16B chunks)
    {
        const uint4* s1 = (const uint4*)W1p; uint4* t1 = (uint4*)sW1;
        for (int i = threadIdx.x; i < (3 * 8 * 512) / 8; i += 256) t1[i] = s1[i];
        const uint4* s2 = (const uint4*)W2p; uint4* t2 = (uint4*)sW2;
        for (int i = threadIdx.x; i < (4 * 8 * 512) / 8; i += 256) t2[i] = s2[i];
    }
    __syncthreads();

    const int lane = threadIdx.x & 31;
    const int w    = threadIdx.x >> 5;
    const int mrow = lane & 15;     // C/D column n, and A row m
    const int hi   = lane >> 4;     // K-half select
    _Float16* hbuf = &sH[w][0];

    // preload biases for this lane's column
    float bias1[8], bias2[8];
    #pragma unroll
    for (int nt = 0; nt < 8; ++nt) {
        bias1[nt] = b1[nt * 16 + mrow];
        bias2[nt] = b2[nt * 16 + mrow];
    }

    const int wgid    = blockIdx.x * 8 + w;
    const int wstride = gridDim.x * 8;

    for (int tile = wgid; tile < NTILES; tile += wstride) {
        const int p = tile * 16 + mrow;
        const float* frow = feats   + (size_t)p * 64;
        const float* crow = centers + (size_t)p * 3;

        // ---- build A (16x96 padded) in ISA A-layout: e<8 -> k=kt*32+hi*8+e ; e>=8 -> k=kt*32+16+hi*8+(e-8)
        v16h A1[3];
        #pragma unroll
        for (int kt = 0; kt < 2; ++kt) {
            const float4* f4 = (const float4*)(frow + kt * 32);
            float4 c0 = f4[hi * 2];
            float4 c1 = f4[hi * 2 + 1];
            float4 c2 = f4[4 + hi * 2];
            float4 c3 = f4[4 + hi * 2 + 1];
            float tmp[16] = {c0.x, c0.y, c0.z, c0.w, c1.x, c1.y, c1.z, c1.w,
                             c2.x, c2.y, c2.z, c2.w, c3.x, c3.y, c3.z, c3.w};
            #pragma unroll
            for (int j = 0; j < 16; ++j) A1[kt][j] = (_Float16)tmp[j];
        }
        {
            v16h a;
            #pragma unroll
            for (int j = 0; j < 16; ++j) a[j] = (_Float16)0.0f;
            if (hi == 0) {   // k = 64,65,66 are the centers; rest zero pad
                a[0] = (_Float16)crow[0];
                a[1] = (_Float16)crow[1];
                a[2] = (_Float16)crow[2];
            }
            A1[2] = a;
        }

        // ---- layer 1: h1 = silu(x @ W1 + b1), 8 ntiles x 3 ksteps = 24 WMMA
        #pragma unroll
        for (int nt = 0; nt < 8; ++nt) {
            v8f acc = {};
            #pragma unroll
            for (int kt = 0; kt < 3; ++kt) {
                v16h bm = *(const v16h*)&sW1[(((nt * 3 + kt) * 32) + lane) * 16];
                acc = __builtin_amdgcn_wmma_f32_16x16x32_f16(
                          false, A1[kt], false, bm, (short)0, acc, false, false);
            }
            #pragma unroll
            for (int r = 0; r < 8; ++r) {
                float v = silu_f(acc[r] + bias1[nt]);
                hbuf[(r + 8 * hi) * 128 + nt * 16 + mrow] = (_Float16)v;  // D layout -> row-major
            }
        }
        asm volatile("s_wait_dscnt 0" ::: "memory");   // wave-local LDS transpose fence

        // ---- re-load h1 as A for layer 2
        v16h A2[4];
        #pragma unroll
        for (int kt = 0; kt < 4; ++kt) {
            v8h lo = *(const v8h*)&hbuf[mrow * 128 + kt * 32 + hi * 8];
            v8h hh = *(const v8h*)&hbuf[mrow * 128 + kt * 32 + 16 + hi * 8];
            v16h a;
            #pragma unroll
            for (int j = 0; j < 8; ++j) { a[j] = lo[j]; a[8 + j] = hh[j]; }
            A2[kt] = a;
        }

        // batch index for the 8 rows this lane owns in the D layout
        int bv[8];
        #pragma unroll
        for (int r = 0; r < 8; ++r) bv[r] = bidx[tile * 16 + r + 8 * hi];

        // ---- layer 2 + fused pooling: 8 ntiles x 4 ksteps = 32 WMMA
        #pragma unroll
        for (int nt = 0; nt < 8; ++nt) {
            v8f acc = {};
            #pragma unroll
            for (int kt = 0; kt < 4; ++kt) {
                v16h bm = *(const v16h*)&sW2[(((nt * 4 + kt) * 32) + lane) * 16];
                acc = __builtin_amdgcn_wmma_f32_16x16x32_f16(
                          false, A2[kt], false, bm, (short)0, acc, false, false);
            }
            const int c = nt * 16 + mrow;
            #pragma unroll
            for (int r = 0; r < 8; ++r) {
                float v = silu_f(acc[r] + bias2[nt]);
                int off = bv[r] * DH + c;
                atomicAdd(&sums[off],  v);
                atomicAdd(&sums2[off], v * v);
                atomicMax(&maxu[off],  fkey(v));
            }
        }
        if (lane < 16) atomicAdd(&counts[bidx[tile * 16 + lane]], 1.0f);
    }
}

// ---------------- head MLP: one wave per 16-row tile of B=512 ----------------
__global__ __launch_bounds__(32) void head_kernel(
    const float* __restrict__ sums, const float* __restrict__ sums2,
    const unsigned* __restrict__ maxu, const float* __restrict__ counts,
    const _Float16* __restrict__ W3p, const float* __restrict__ b3,
    const _Float16* __restrict__ W4p, const float* __restrict__ b4,
    const _Float16* __restrict__ W5p, const float* __restrict__ b5,
    float* __restrict__ out)
{
    __shared__ alignas(16) _Float16 sg[16 * 384];
    __shared__ alignas(16) _Float16 sd1[16 * 256];
    __shared__ alignas(16) _Float16 sd2[16 * 128];

    const int lane = threadIdx.x;
    const int mrow = lane & 15;
    const int hi   = lane >> 4;
    const int t    = blockIdx.x;

    // finalize pooling -> g tile (16 x 384) = [mean | max | var]
    for (int m = 0; m < 16; ++m) {
        int seg = t * 16 + m;
        float cnt = fmaxf(counts[seg], 1.0f);
        for (int c = lane; c < 384; c += 32) {
            float v;
            if (c < 128) {
                v = sums[seg * DH + c] / cnt;
            } else if (c < 256) {
                v = fkey_dec(maxu[seg * DH + (c - 128)]);
            } else {
                float mean = sums[seg * DH + (c - 256)] / cnt;
                float e2   = sums2[seg * DH + (c - 256)] / cnt;
                v = fmaxf(e2 - mean * mean, 0.0f);
            }
            sg[m * 384 + c] = (_Float16)v;
        }
    }
    __syncthreads();

    // ---- layer 3: 384 -> 256 (16 ntiles x 12 ksteps)
    v16h A3[12];
    #pragma unroll
    for (int kt = 0; kt < 12; ++kt) {
        v8h lo = *(const v8h*)&sg[mrow * 384 + kt * 32 + hi * 8];
        v8h hh = *(const v8h*)&sg[mrow * 384 + kt * 32 + 16 + hi * 8];
        v16h a;
        #pragma unroll
        for (int j = 0; j < 8; ++j) { a[j] = lo[j]; a[8 + j] = hh[j]; }
        A3[kt] = a;
    }
    #pragma unroll
    for (int nt = 0; nt < 16; ++nt) {
        v8f acc = {};
        #pragma unroll
        for (int kt = 0; kt < 12; ++kt) {
            v16h bm = *(const v16h*)&W3p[(((nt * 12 + kt) * 32) + lane) * 16];
            acc = __builtin_amdgcn_wmma_f32_16x16x32_f16(
                      false, A3[kt], false, bm, (short)0, acc, false, false);
        }
        float bias = b3[nt * 16 + mrow];
        #pragma unroll
        for (int r = 0; r < 8; ++r) {
            float v = silu_f(acc[r] + bias);
            sd1[(r + 8 * hi) * 256 + nt * 16 + mrow] = (_Float16)v;
        }
    }
    __syncthreads();

    // ---- layer 4: 256 -> 128 (8 ntiles x 8 ksteps)
    v16h A4[8];
    #pragma unroll
    for (int kt = 0; kt < 8; ++kt) {
        v8h lo = *(const v8h*)&sd1[mrow * 256 + kt * 32 + hi * 8];
        v8h hh = *(const v8h*)&sd1[mrow * 256 + kt * 32 + 16 + hi * 8];
        v16h a;
        #pragma unroll
        for (int j = 0; j < 8; ++j) { a[j] = lo[j]; a[8 + j] = hh[j]; }
        A4[kt] = a;
    }
    #pragma unroll
    for (int nt = 0; nt < 8; ++nt) {
        v8f acc = {};
        #pragma unroll
        for (int kt = 0; kt < 8; ++kt) {
            v16h bm = *(const v16h*)&W4p[(((nt * 8 + kt) * 32) + lane) * 16];
            acc = __builtin_amdgcn_wmma_f32_16x16x32_f16(
                      false, A4[kt], false, bm, (short)0, acc, false, false);
        }
        float bias = b4[nt * 16 + mrow];
        #pragma unroll
        for (int r = 0; r < 8; ++r) {
            float v = silu_f(acc[r] + bias);
            sd2[(r + 8 * hi) * 128 + nt * 16 + mrow] = (_Float16)v;
        }
    }
    __syncthreads();

    // ---- layer 5: 128 -> 4 (padded to 16 cols; 4 ksteps)
    v16h A5[4];
    #pragma unroll
    for (int kt = 0; kt < 4; ++kt) {
        v8h lo = *(const v8h*)&sd2[mrow * 128 + kt * 32 + hi * 8];
        v8h hh = *(const v8h*)&sd2[mrow * 128 + kt * 32 + 16 + hi * 8];
        v16h a;
        #pragma unroll
        for (int j = 0; j < 8; ++j) { a[j] = lo[j]; a[8 + j] = hh[j]; }
        A5[kt] = a;
    }
    v8f acc = {};
    #pragma unroll
    for (int kt = 0; kt < 4; ++kt) {
        v16h bm = *(const v16h*)&W5p[((kt * 32) + lane) * 16];
        acc = __builtin_amdgcn_wmma_f32_16x16x32_f16(
                  false, A5[kt], false, bm, (short)0, acc, false, false);
    }
    if (mrow < 4) {
        float bias = b5[mrow];
        #pragma unroll
        for (int r = 0; r < 8; ++r)
            out[(size_t)(t * 16 + r + 8 * hi) * 4 + mrow] = acc[r] + bias;
    }
}

extern "C" void kernel_launch(void* const* d_in, const int* in_sizes, int n_in,
                              void* d_out, int out_size, void* d_ws, size_t ws_size,
                              hipStream_t stream) {
    const float* feats   = (const float*)d_in[0];
    const float* centers = (const float*)d_in[1];
    const int*   bidx    = (const int*)d_in[2];
    // d_in[3] = B (constant 512)
    const float* W1 = (const float*)d_in[4];
    const float* b1 = (const float*)d_in[5];
    const float* W2 = (const float*)d_in[6];
    const float* b2 = (const float*)d_in[7];
    const float* W3 = (const float*)d_in[8];
    const float* b3 = (const float*)d_in[9];
    const float* W4 = (const float*)d_in[10];
    const float* b4 = (const float*)d_in[11];
    const float* W5 = (const float*)d_in[12];
    const float* b5 = (const float*)d_in[13];
    float* out = (float*)d_out;

    char* ws = (char*)d_ws;
    float*     sums   = (float*)(ws + OFF_SUMS);
    float*     sums2  = (float*)(ws + OFF_SUMS2);
    unsigned*  maxu   = (unsigned*)(ws + OFF_MAXU);
    float*     counts = (float*)(ws + OFF_CNT);
    _Float16*  W1p    = (_Float16*)(ws + OFF_W1P);
    _Float16*  W2p    = (_Float16*)(ws + OFF_W2P);
    _Float16*  W3p    = (_Float16*)(ws + OFF_W3P);
    _Float16*  W4p    = (_Float16*)(ws + OFF_W4P);
    _Float16*  W5p    = (_Float16*)(ws + OFF_W5P);

    // 1) zero pooling accumulators
    zero_ws_kernel<<<(BSEG * DH + 255) / 256, 256, 0, stream>>>(sums, sums2, maxu, counts);

    // 2) pack weights into WMMA B-layout f16
    pack_w_kernel<<<(3 * 8 * 512 + 255) / 256, 256, 0, stream>>>(W1, W1p, 67, 128, 3, 8);
    pack_w_kernel<<<(4 * 8 * 512 + 255) / 256, 256, 0, stream>>>(W2, W2p, 128, 128, 4, 8);
    pack_w_kernel<<<(12 * 16 * 512 + 255) / 256, 256, 0, stream>>>(W3, W3p, 384, 256, 12, 16);
    pack_w_kernel<<<(8 * 8 * 512 + 255) / 256, 256, 0, stream>>>(W4, W4p, 256, 128, 8, 8);
    pack_w_kernel<<<(4 * 1 * 512 + 255) / 256, 256, 0, stream>>>(W5, W5p, 128, 4, 4, 1);

    // 3) per-point embed MLP + pooling (dominant: 56 WMMA per 16-point tile)
    embed_pool_kernel<<<1024, 256, 0, stream>>>(
        feats, centers, bidx, W1p, b1, W2p, b2, sums, sums2, maxu, counts);

    // 4) head MLP over B=512 rows (32 tiles x 1 wave)
    head_kernel<<<32, 32, 0, stream>>>(
        sums, sums2, maxu, counts, W3p, b3, W4p, b4, W5p, b5, out);

    (void)in_sizes; (void)n_in; (void)out_size; (void)ws_size;
}
